// GVPMultiEdgeConv_2585570312764
// MI455X (gfx1250) — matrix operand order repaired and checked by reading, not compiled, not run on hardware
//
#include <hip/hip_runtime.h>
#include <hip/hip_bf16.h>
#include <cstdint>

// GVP multi-edge conv for MI455X (gfx1250, wave32, WMMA + async-to-LDS).
// Per-16-edge/node workgroup: stage 161/144-wide scalar rows as bf16 in LDS,
// double-buffer the packed bf16 Wf weight tiles into LDS with
// GLOBAL_LOAD_ASYNC_TO_LDS_B128 (ASYNCcnt + barrier pipeline), and run
// v_wmma_f32_16x16x32_bf16.  Vector-channel matmuls / RBF / norms / gating in
// VALU; scatter via global f32 atomics (L2-resident node tables).

#define SCALAR 128
#define VEC 16

typedef __attribute__((ext_vector_type(16))) __bf16 v16bf;
typedef __attribute__((ext_vector_type(8)))  __bf16 v8bf;
typedef __attribute__((ext_vector_type(4)))  __bf16 v4bf;
typedef __attribute__((ext_vector_type(8)))  float  v8f;

// GCC-style vector types to match the async-load builtin's parameter types
// (diagnostic showed: "__vector_size__(4 * sizeof(int)) int __device__ *").
typedef int v4i_ __attribute__((vector_size(16)));
typedef __attribute__((address_space(1))) v4i_ as1_v4i;
typedef __attribute__((address_space(3))) v4i_ as3_v4i;

#if defined(__HIP_DEVICE_COMPILE__) && __has_builtin(__builtin_amdgcn_global_load_async_to_lds_b128)
#define HAVE_ASYNC_LDS 1
#else
#define HAVE_ASYNC_LDS 0
#endif

__device__ __forceinline__ float sigmoidf_(float x){
  return __builtin_amdgcn_rcpf(1.0f + __expf(-x));   // v_rcp_f32, not full div
}
__device__ __forceinline__ float siluf_(float x){ return x * sigmoidf_(x); }

// ---------------- utility kernels ----------------

__global__ void zero_kernel(float* p, long long n){
  long long i = (long long)blockIdx.x*256 + threadIdx.x;
  if (i < n) p[i] = 0.0f;
}

// Repack Wf [K,128] f32 (row-major) -> bf16 [Kpad/16][128][16]:
// dst[(k/16)][col][k%16].  A lane's 16 B-fragment values are then contiguous.
__global__ void cvt_wf_kernel(const float* __restrict__ src, __bf16* __restrict__ dst,
                              int K, int Kpad){
  int i = blockIdx.x*256 + threadIdx.x;
  if (i >= Kpad*128) return;
  int k = i >> 7, c = i & 127;
  float v = (k < K) ? src[k*128 + c] : 0.0f;
  dst[((k >> 4) << 11) + (c << 4) + (k & 15)] = (__bf16)v;
}

// ---------------- async B-tile staging ----------------
// One K-tile of B = 2 groups * 2048 bf16 = 8KB.  256 threads copy 2x16B each.
__device__ __forceinline__ void stage_tile(const __bf16* __restrict__ src,
                                           __bf16* dst, int tid){
#if HAVE_ASYNC_LDS
  #pragma unroll
  for (int i = 0; i < 2; ++i){
    int o = (tid + i*256) * 8;                       // 8 bf16 = 16B per lane
    __builtin_amdgcn_global_load_async_to_lds_b128(
        (as1_v4i*)(src + o), (as3_v4i*)(dst + o), 0, 0);
  }
#else
  #pragma unroll
  for (int i = 0; i < 2; ++i){
    int o = (tid + i*256) * 8;
    *reinterpret_cast<uint4*>(dst + o) = *reinterpret_cast<const uint4*>(src + o);
  }
#endif
}

__device__ __forceinline__ void wait_async0(){
#if HAVE_ASYNC_LDS
 #if __has_builtin(__builtin_amdgcn_s_wait_asynccnt)
  __builtin_amdgcn_s_wait_asynccnt(0);
 #else
  asm volatile("s_wait_asynccnt 0x0" ::: "memory");
 #endif
#endif
}

// ---------------- pipelined WMMA GEMM ----------------
// A: LDS bf16 [16][ROWLEN], rows = M.  B: packed global, staged tile-by-tile
// into sWB[2][4096] (double buffer).  Wave owns 16 output columns.
template<int KTILES, int ROWLEN>
__device__ __forceinline__ v8f gemm_bf16_pipe(const __bf16 (*sA)[ROWLEN],
                                              const __bf16* __restrict__ WB,
                                              __bf16* sWB, int tid){
  const int lane = tid & 31, wave = tid >> 5;
  const int mrow = lane & 15;
  const int hi   = lane >> 4;              // which K-half this lane holds
  const int col  = (wave << 4) + mrow;
  v8f acc = {};
  stage_tile(WB, sWB, tid);                // prologue: tile 0 -> buf 0
  #pragma unroll
  for (int kt = 0; kt < KTILES; ++kt){
    wait_async0();                         // own copies for tile kt done
    __syncthreads();                       // everyone's copies visible
    if (kt + 1 < KTILES)                   // prefetch next tile into other buf
      stage_tile(WB + (kt + 1)*4096, sWB + ((kt + 1) & 1)*4096, tid);
    // A fragment (16-bit 16x32): lanes 0-15 rows, K {0..15};
    // lanes 16-31 hold K {16..31}.
    const int ka = kt*32 + hi*16;
    v8bf a_lo = *reinterpret_cast<const v8bf*>(&sA[mrow][ka]);
    v8bf a_hi = *reinterpret_cast<const v8bf*>(&sA[mrow][ka + 8]);
    v16bf a;
    #pragma unroll
    for (int i = 0; i < 8; ++i){ a[i] = a_lo[i]; a[8+i] = a_hi[i]; }
    // B fragment from staged LDS tile.
    const __bf16* wp = sWB + ((kt & 1) << 12) + (hi << 11) + (col << 4);
    v8bf b_lo = *reinterpret_cast<const v8bf*>(wp);
    v8bf b_hi = *reinterpret_cast<const v8bf*>(wp + 8);
    v16bf b;
    #pragma unroll
    for (int i = 0; i < 8; ++i){ b[i] = b_lo[i]; b[8+i] = b_hi[i]; }
    acc = __builtin_amdgcn_wmma_f32_16x16x32_bf16(false, a, false, b,
                                                  (short)0, acc, false, false);
  }
  return acc;
}

// ---------------- edge message kernel ----------------
__global__ __launch_bounds__(256, 2)
void edge_msg_kernel(const float* __restrict__ h_src,   // [Ns,128]
                     const float* __restrict__ v_src,   // [Ns,16,3]
                     const float* __restrict__ x_src,   // [Ns,3]
                     const float* __restrict__ x_dst,   // [Nd,3]
                     const int*   __restrict__ src,
                     const int*   __restrict__ dst,
                     const float* __restrict__ Wh,      // [17*17] (v,h)
                     const float* __restrict__ Wu,      // [17*16] (h,u)
                     const __bf16* __restrict__ WfB,    // packed [12][128][16]
                     const float* __restrict__ bfv,     // [128]
                     const float* __restrict__ Wg,      // [128*16]
                     const float* __restrict__ bg,      // [16]
                     float* __restrict__ sm_acc,        // [Nd*128]
                     float* __restrict__ vm_acc,        // [Nd*48]
                     int E)
{
  __shared__ __align__(16) __bf16 sWB[2*4096];  // 16KB double buffer for B
  __shared__ __align__(16) __bf16 sA[16][192];  // [h(128)|rbf(16)|sh(17)|pad]
  __shared__ float sVh[16][17*3];
  __shared__ float sVu[16][48];
  __shared__ float sSo[16][128];
  __shared__ float sDir[16][3];
  __shared__ float sDij[16];
  __shared__ int   sSrc[16], sDst[16], sValid[16];

  const int tid = threadIdx.x;
  const int r   = tid >> 4;     // edge row 0..15
  const int cl  = tid & 15;
  const int e0  = blockIdx.x << 4;

  if (tid < 16){
    int e = e0 + tid;
    int ok = (e < E);
    sValid[tid] = ok;
    int ec = ok ? e : 0;
    sSrc[tid] = src[ec];
    sDst[tid] = dst[ec];
  }
  __syncthreads();

  // geometry
  if (tid < 16){
    int si = sSrc[tid], di = sDst[tid];
    float dx = x_src[si*3+0] - x_dst[di*3+0];
    float dy = x_src[si*3+1] - x_dst[di*3+1];
    float dz = x_src[si*3+2] - x_dst[di*3+2];
    float d2 = fmaxf(dx*dx + dy*dy + dz*dz, 1e-8f);
    float d  = sqrtf(d2) + 1e-8f;
    sDij[tid] = d;
    float inv = 1.0f/d;
    sDir[tid][0] = dx*inv; sDir[tid][1] = dy*inv; sDir[tid][2] = dz*inv;
  }
  // gather h -> A cols [0,128): float4 loads, packed v4bf (ds_store_b64)
  {
    const float4* hp4 = (const float4*)(h_src + (size_t)sSrc[r]*SCALAR);
    #pragma unroll
    for (int j = 0; j < 2; ++j){
      float4 f = hp4[cl + (j<<4)];
      v4bf p; p[0]=(__bf16)f.x; p[1]=(__bf16)f.y; p[2]=(__bf16)f.z; p[3]=(__bf16)f.w;
      *reinterpret_cast<v4bf*>(&sA[r][(cl + (j<<4)) << 2]) = p;
    }
  }
  // zero A cols [144,192) with b128 stores (sh overwrites [144,161) later)
  for (int t = tid; t < 16*6; t += 256){
    uint4 z = {0,0,0,0};
    *reinterpret_cast<uint4*>(&sA[t/6][144 + (t % 6)*8]) = z;
  }
  __syncthreads();

  // RBF -> A cols [128,144): mu_j = j, sigma = 15/16
  {
    float t = (sDij[r] - (float)cl) * (16.0f/15.0f);
    sA[r][128 + cl] = (__bf16)__expf(-t*t);
  }
  // Vh[h][c] = sum_v vin[v][c]*Wh[v][h], vin = [v_src[src] ; x_dir]
  for (int t = tid; t < 16*51; t += 256){
    int e = t/51, idx = t - e*51, hh = idx/3, c = idx - hh*3;
    const float* vp = v_src + (size_t)sSrc[e]*48;
    float acc = sDir[e][c] * Wh[16*17 + hh];
    #pragma unroll
    for (int v = 0; v < 16; ++v) acc += vp[v*3 + c] * Wh[v*17 + hh];
    sVh[e][hh*3 + c] = acc;
  }
  __syncthreads();

  // sh -> A cols [144,161)
  for (int t = tid; t < 16*17; t += 256){
    int e = t/17, hh = t - e*17;
    const float* q = &sVh[e][hh*3];
    float n2 = fmaxf(q[0]*q[0] + q[1]*q[1] + q[2]*q[2], 1e-8f);
    sA[e][144 + hh] = (__bf16)sqrtf(n2);
  }
  // Vu[u][c] = sum_h Vh[h][c]*Wu[h][u]
  for (int t = tid; t < 16*48; t += 256){
    int e = t/48, idx = t - e*48, u = idx/3, c = idx - u*3;
    float acc = 0.0f;
    #pragma unroll
    for (int hh = 0; hh < 17; ++hh) acc += sVh[e][hh*3 + c] * Wu[hh*16 + u];
    sVu[e][idx] = acc;
  }
  __syncthreads();

  // so = silu(A[16x192] @ Wf[192x128] + bf)  via pipelined WMMA
  {
    v8f acc = gemm_bf16_pipe<6, 192>(sA, WfB, sWB, tid);
    int lane = tid & 31, wave = tid >> 5;
    int mrow = lane & 15, hi = lane >> 4;
    int col = (wave << 4) + mrow;
    float bias = bfv[col];
    #pragma unroll
    for (int rr = 0; rr < 8; ++rr){
      int row = rr + hi*8;                 // C layout: vgpr r -> M=r / M=8+r
      sSo[row][col] = siluf_(acc[rr] + bias);
    }
  }
  __syncthreads();

  // gate = sigmoid(so @ Wg + bg); scatter so and gate*Vu
  if (sValid[r]){
    float g = bg[cl];
    for (int k = 0; k < 128; ++k) g += sSo[r][k] * Wg[k*16 + cl];
    g = sigmoidf_(g);
    int dn = sDst[r];
    float* vmp = vm_acc + (size_t)dn*48 + cl*3;
    atomicAdd(&vmp[0], g * sVu[r][cl*3 + 0]);
    atomicAdd(&vmp[1], g * sVu[r][cl*3 + 1]);
    atomicAdd(&vmp[2], g * sVu[r][cl*3 + 2]);
    float* smp = sm_acc + (size_t)dn*SCALAR;
    #pragma unroll
    for (int j = 0; j < 8; ++j)
      atomicAdd(&smp[cl + (j<<4)], sSo[r][cl + (j<<4)]);
  }
}

// ---------------- node update kernel ----------------
__global__ __launch_bounds__(256, 2)
void node_update_kernel(const float* __restrict__ h_in,    // [N,128]
                        const float* __restrict__ v_in,    // [N,16,3]
                        const float* __restrict__ mln_w, const float* __restrict__ mln_b,
                        const float* __restrict__ uln_w, const float* __restrict__ uln_b,
                        const float* __restrict__ Wh,      // [16*16]
                        const float* __restrict__ Wu,      // [16*16]
                        const __bf16* __restrict__ WfB,    // packed [10][128][16]
                        const float* __restrict__ bfv,
                        const float* __restrict__ Wg,
                        const float* __restrict__ bg,
                        float* __restrict__ s_io,          // [N*128] acc in / out
                        float* __restrict__ v_io,          // [N*48]
                        int N)
{
  __shared__ __align__(16) __bf16 sWB[2*4096];
  __shared__ float sS[16][128];
  __shared__ float sV[16][48];
  __shared__ __align__(16) __bf16 sA[16][160];  // [sn(128)|sh(16)|pad(16)]
  __shared__ float sVh[16][48];
  __shared__ float sVu[16][48];
  __shared__ float sDs[16][128];
  __shared__ float sMu[16], sRstd[16], sVn[16];

  const int tid = threadIdx.x;
  const int r = tid >> 4, cl = tid & 15;
  const int n = (blockIdx.x << 4) + r;       // N % 16 == 0 here
  (void)N;

  // s = h + sm/10 ; v = v + vm/10
  #pragma unroll
  for (int j = 0; j < 8; ++j){
    int c = cl + (j<<4);
    sS[r][c] = h_in[(size_t)n*128 + c] + s_io[(size_t)n*128 + c]*0.1f;
  }
  #pragma unroll
  for (int c = 0; c < 3; ++c){
    int idx = cl*3 + c;
    sV[r][idx] = v_in[(size_t)n*48 + idx] + v_io[(size_t)n*48 + idx]*0.1f;
  }
  __syncthreads();

  // mln stats
  if (tid < 16){
    float mu = 0.0f;
    for (int k = 0; k < 128; ++k) mu += sS[tid][k];
    mu *= (1.0f/128.0f);
    float var = 0.0f;
    for (int k = 0; k < 128; ++k){ float d = sS[tid][k]-mu; var += d*d; }
    var *= (1.0f/128.0f);
    sMu[tid] = mu; sRstd[tid] = rsqrtf(var + 1e-5f);
    float acc = 0.0f;
    for (int v = 0; v < 16; ++v){
      const float* q = &sV[tid][v*3];
      acc += fmaxf(q[0]*q[0]+q[1]*q[1]+q[2]*q[2], 1e-8f);
    }
    sVn[tid] = sqrtf(acc * (1.0f/16.0f));
  }
  __syncthreads();

  {
    float mu = sMu[r], rstd = sRstd[r], invVn = __builtin_amdgcn_rcpf(sVn[r]);
    #pragma unroll
    for (int j = 0; j < 8; ++j){
      int c = cl + (j<<4);
      float sn = (sS[r][c]-mu)*rstd*mln_w[c] + mln_b[c];
      sS[r][c] = sn;
      sA[r][c] = (__bf16)sn;
    }
    #pragma unroll
    for (int c = 0; c < 3; ++c) sV[r][cl*3 + c] *= invVn;
  }
  // zero pad cols [144,160)
  if (tid < 16*2){
    uint4 z = {0,0,0,0};
    *reinterpret_cast<uint4*>(&sA[tid>>1][144 + (tid&1)*8]) = z;
  }
  __syncthreads();

  // Vh = vv @ Wh
  for (int t = tid; t < 16*48; t += 256){
    int e = t/48, idx = t - e*48, hh = idx/3, c = idx - hh*3;
    float acc = 0.0f;
    #pragma unroll
    for (int v = 0; v < 16; ++v) acc += sV[e][v*3 + c] * Wh[v*16 + hh];
    sVh[e][idx] = acc;
  }
  __syncthreads();

  // sh -> A cols [128,144)
  {
    const float* q = &sVh[r][cl*3];
    float n2 = fmaxf(q[0]*q[0]+q[1]*q[1]+q[2]*q[2], 1e-8f);
    sA[r][128 + cl] = (__bf16)sqrtf(n2);
  }
  // Vu = Vh @ Wu
  for (int t = tid; t < 16*48; t += 256){
    int e = t/48, idx = t - e*48, u = idx/3, c = idx - u*3;
    float acc = 0.0f;
    #pragma unroll
    for (int hh = 0; hh < 16; ++hh) acc += sVh[e][hh*3 + c] * Wu[hh*16 + u];
    sVu[e][idx] = acc;
  }
  __syncthreads();

  // ds = silu(A[16x160] @ Wf[160x128] + bf)
  {
    v8f acc = gemm_bf16_pipe<5, 160>(sA, WfB, sWB, tid);
    int lane = tid & 31, wave = tid >> 5;
    int mrow = lane & 15, hi = lane >> 4;
    int col = (wave << 4) + mrow;
    float bias = bfv[col];
    #pragma unroll
    for (int rr = 0; rr < 8; ++rr)
      sDs[rr + hi*8][col] = siluf_(acc[rr] + bias);
  }
  __syncthreads();

  // gate; v2 = vv + sigmoid(gate)*Vu (in place on sV)
  {
    float g = bg[cl];
    for (int k = 0; k < 128; ++k) g += sDs[r][k] * Wg[k*16 + cl];
    g = sigmoidf_(g);
    #pragma unroll
    for (int c = 0; c < 3; ++c) sV[r][cl*3 + c] += g * sVu[r][cl*3 + c];
  }
  __syncthreads();

  // uln stats on s2 = sS + sDs, v2 = sV
  if (tid < 16){
    float mu = 0.0f;
    for (int k = 0; k < 128; ++k) mu += sS[tid][k] + sDs[tid][k];
    mu *= (1.0f/128.0f);
    float var = 0.0f;
    for (int k = 0; k < 128; ++k){ float d = sS[tid][k]+sDs[tid][k]-mu; var += d*d; }
    var *= (1.0f/128.0f);
    sMu[tid] = mu; sRstd[tid] = rsqrtf(var + 1e-5f);
    float acc = 0.0f;
    for (int v = 0; v < 16; ++v){
      const float* q = &sV[tid][v*3];
      acc += fmaxf(q[0]*q[0]+q[1]*q[1]+q[2]*q[2], 1e-8f);
    }
    sVn[tid] = sqrtf(acc * (1.0f/16.0f));
  }
  __syncthreads();

  {
    float mu = sMu[r], rstd = sRstd[r], invVn = __builtin_amdgcn_rcpf(sVn[r]);
    #pragma unroll
    for (int j = 0; j < 8; ++j){
      int c = cl + (j<<4);
      float s2 = sS[r][c] + sDs[r][c];
      s_io[(size_t)n*128 + c] = (s2-mu)*rstd*uln_w[c] + uln_b[c];
    }
    #pragma unroll
    for (int c = 0; c < 3; ++c)
      v_io[(size_t)n*48 + cl*3 + c] = sV[r][cl*3 + c] * invVn;
  }
}

// ---------------- launch ----------------
// Params flatten (JAX dict = alphabetical):
// 12..15: mln.lig{b,w}, mln.rec{b,w}
// 16..21: msg.ll{Wf,Wg,Wh,Wu,bf,bg}; 22..27 msg.rl; 28..33 msg.rr
// 34..37: uln.lig{b,w}, uln.rec{b,w}
// 38..43: upd.lig{Wf,Wg,Wh,Wu,bf,bg}; 44..49 upd.rec
extern "C" void kernel_launch(void* const* d_in, const int* in_sizes, int n_in,
                              void* d_out, int out_size, void* d_ws, size_t ws_size,
                              hipStream_t stream)
{
  const float* lig_h = (const float*)d_in[0];
  const float* lig_x = (const float*)d_in[1];
  const float* lig_v = (const float*)d_in[2];
  const float* rec_h = (const float*)d_in[3];
  const float* rec_x = (const float*)d_in[4];
  const float* rec_v = (const float*)d_in[5];
  const int* ll_src = (const int*)d_in[6];
  const int* ll_dst = (const int*)d_in[7];
  const int* rl_src = (const int*)d_in[8];
  const int* rl_dst = (const int*)d_in[9];
  const int* rr_src = (const int*)d_in[10];
  const int* rr_dst = (const int*)d_in[11];

  const int NL = in_sizes[0] / SCALAR;
  const int NR = in_sizes[3] / SCALAR;

  float* out    = (float*)d_out;
  float* lig_s  = out;                              // [NL,128] acc -> out
  float* lig_vv = lig_s  + (size_t)NL*SCALAR;       // [NL,48]
  float* rec_s  = lig_vv + (size_t)NL*48;           // [NR,128]
  float* rec_vv = rec_s  + (size_t)NR*SCALAR;       // [NR,48]

  __bf16* ws   = (__bf16*)d_ws;
  __bf16* WfLL = ws;                 // 192*128
  __bf16* WfRL = ws + 24576;
  __bf16* WfRR = ws + 49152;
  __bf16* WfUL = ws + 73728;         // 160*128
  __bf16* WfUR = ws + 94208;
  (void)ws_size; (void)n_in;

  // 1) zero accumulators (entire d_out)
  long long tot = (long long)out_size;
  zero_kernel<<<(unsigned)((tot + 255) / 256), 256, 0, stream>>>(out, tot);

  // 2) repack Wf weights to fragment-friendly bf16
  cvt_wf_kernel<<<96, 256, 0, stream>>>((const float*)d_in[16], WfLL, 161, 192);
  cvt_wf_kernel<<<96, 256, 0, stream>>>((const float*)d_in[22], WfRL, 161, 192);
  cvt_wf_kernel<<<96, 256, 0, stream>>>((const float*)d_in[28], WfRR, 161, 192);
  cvt_wf_kernel<<<80, 256, 0, stream>>>((const float*)d_in[38], WfUL, 144, 160);
  cvt_wf_kernel<<<80, 256, 0, stream>>>((const float*)d_in[44], WfUR, 144, 160);

  // 3) edge messages (+ atomic scatter into accumulators)
  {
    int E = in_sizes[6];
    edge_msg_kernel<<<(E + 15)/16, 256, 0, stream>>>(
        lig_h, lig_v, lig_x, lig_x, ll_src, ll_dst,
        (const float*)d_in[18], (const float*)d_in[19], WfLL,
        (const float*)d_in[20], (const float*)d_in[17], (const float*)d_in[21],
        lig_s, lig_vv, E);
  }
  {
    int E = in_sizes[8];
    edge_msg_kernel<<<(E + 15)/16, 256, 0, stream>>>(
        rec_h, rec_v, rec_x, lig_x, rl_src, rl_dst,
        (const float*)d_in[24], (const float*)d_in[25], WfRL,
        (const float*)d_in[26], (const float*)d_in[23], (const float*)d_in[27],
        lig_s, lig_vv, E);
  }
  {
    int E = in_sizes[10];
    edge_msg_kernel<<<(E + 15)/16, 256, 0, stream>>>(
        rec_h, rec_v, rec_x, rec_x, rr_src, rr_dst,
        (const float*)d_in[30], (const float*)d_in[31], WfRR,
        (const float*)d_in[32], (const float*)d_in[29], (const float*)d_in[33],
        rec_s, rec_vv, E);
  }

  // 4) node updates (read accumulators, write final outputs in place)
  node_update_kernel<<<NL/16, 256, 0, stream>>>(
      lig_h, lig_v,
      (const float*)d_in[13], (const float*)d_in[12],
      (const float*)d_in[35], (const float*)d_in[34],
      (const float*)d_in[40], (const float*)d_in[41], WfUL,
      (const float*)d_in[42], (const float*)d_in[39], (const float*)d_in[43],
      lig_s, lig_vv, NL);
  node_update_kernel<<<NR/16, 256, 0, stream>>>(
      rec_h, rec_v,
      (const float*)d_in[15], (const float*)d_in[14],
      (const float*)d_in[37], (const float*)d_in[36],
      (const float*)d_in[46], (const float*)d_in[47], WfUR,
      (const float*)d_in[48], (const float*)d_in[45], (const float*)d_in[49],
      rec_s, rec_vv, NR);
}
